// Distortion_47253230191385
// MI455X (gfx1250) — compile-verified, matrix operand order checked by reading, not run on hardware
//
#include <hip/hip_runtime.h>
#include <hip/hip_bf16.h>
#include <stdint.h>
#include <stddef.h>

// Problem constants (match reference)
#define NPTS 8192
#define KDIM 128
#define BT   128           // output tile edge per block
#define NT   (NPTS / BT)   // 64 tiles per edge
#define LDK  (KDIM + 8)    // LDS row stride in bf16 (272B, 16B-aligned, bank-rotating)

typedef __bf16 bf16;
typedef __attribute__((ext_vector_type(8)))  __bf16 v8bf;
typedef __attribute__((ext_vector_type(16))) __bf16 v16bf;
typedef __attribute__((ext_vector_type(8)))  float  v8f;

// CDNA5 async global->LDS copy (VGLOBAL GVS form), tracked by ASYNCcnt.
// lds_addr: wave-relative LDS byte address (low 32 bits of flat shared ptr)
// goff:     per-lane 32-bit byte offset added to the 64-bit SGPR base
__device__ __forceinline__ void async_load_b128(uint32_t lds_addr, uint32_t goff,
                                                const void* sbase) {
  asm volatile("global_load_async_to_lds_b128 %0, %1, %2"
               :
               : "v"(lds_addr), "v"(goff), "s"(sbase)
               : "memory");
}
__device__ __forceinline__ void wait_asynccnt0() {
  asm volatile("s_wait_asynccnt 0x0" ::: "memory");
}

__global__ void dist_zero_kernel(float* out) { out[0] = 0.0f; }

// One block per row: write bf16 hi/lo split of mapping + row squared norm.
__global__ __launch_bounds__(KDIM)
void dist_prep_kernel(const float* __restrict__ mapping,
                      float* __restrict__ sq,
                      bf16* __restrict__ hi,
                      bf16* __restrict__ lo)
{
  const int i = blockIdx.x;
  const int k = threadIdx.x;
  const float v = mapping[(size_t)i * KDIM + k];
  const bf16 h = (bf16)v;
  const bf16 l = (bf16)(v - (float)h);
  hi[(size_t)i * KDIM + k] = h;
  lo[(size_t)i * KDIM + k] = l;

  __shared__ float red[KDIM];
  red[k] = v * v;
  __syncthreads();
  for (int s = KDIM / 2; s > 0; s >>= 1) {
    if (k < s) red[k] += red[k + s];
    __syncthreads();
  }
  if (k == 0) sq[i] = red[0];
}

// Fused Gram(WMMA bf16 split) + distance + distortion-ratio + reduction.
// 1D triangular grid: block t -> upper-triangular tile (by, bx), bx >= by.
__global__ __launch_bounds__(256)
void dist_main_kernel(const float* __restrict__ D,
                      const float* __restrict__ sq,
                      const bf16* __restrict__ hi,
                      const bf16* __restrict__ lo,
                      float* __restrict__ out)
{
  // Decode triangular linear index (cheap SALU loop, <= NT iters)
  int t = blockIdx.x;
  int by = 0, rowlen = NT;
  while (t >= rowlen) { t -= rowlen; --rowlen; ++by; }
  const int bx = by + t;

  const int ibase = by * BT;
  const int jbase = bx * BT;

  __shared__ __align__(16) bf16 sAh[BT * LDK];
  __shared__ __align__(16) bf16 sAl[BT * LDK];
  __shared__ __align__(16) bf16 sBh[BT * LDK];
  __shared__ __align__(16) bf16 sBl[BT * LDK];
  __shared__ float sSqA[BT];
  __shared__ float sSqB[BT];
  __shared__ float sRed[8];

  const int tid = threadIdx.x;

  // Async cooperative tile fill: 2048 16-byte chunks per array, 8 chunks/thread.
  {
    const bf16* hiA = hi + (size_t)ibase * KDIM;
    const bf16* loA = lo + (size_t)ibase * KDIM;
    const bf16* hiB = hi + (size_t)jbase * KDIM;
    const bf16* loB = lo + (size_t)jbase * KDIM;
    const uint32_t ah = (uint32_t)(uintptr_t)&sAh[0];
    const uint32_t al = (uint32_t)(uintptr_t)&sAl[0];
    const uint32_t bh = (uint32_t)(uintptr_t)&sBh[0];
    const uint32_t bl = (uint32_t)(uintptr_t)&sBl[0];
#pragma unroll
    for (int tt = 0; tt < 8; ++tt) {
      const int c   = tid + tt * 256;  // 0..2047
      const int row = c >> 4;          // 0..127
      const int kc  = (c & 15) << 3;   // 0..120 step 8 bf16 (16B)
      const uint32_t goff = (uint32_t)(row * KDIM + kc) * (uint32_t)sizeof(bf16);
      const uint32_t loff = (uint32_t)(row * LDK + kc) * (uint32_t)sizeof(bf16);
      async_load_b128(ah + loff, goff, hiA);
      async_load_b128(al + loff, goff, loA);
      async_load_b128(bh + loff, goff, hiB);
      async_load_b128(bl + loff, goff, loB);
    }
  }
  if (tid < BT) sSqA[tid]      = sq[ibase + tid];
  else          sSqB[tid - BT] = sq[jbase + tid - BT];
  wait_asynccnt0();     // our async writes into LDS landed
  __syncthreads();      // everyone's landed

  // 8 waves in a 4(M) x 2(N) grid; each wave owns a 32x64 sub-tile.
  const int lane = tid & 31;
  const int wid  = tid >> 5;
  const int wm   = wid >> 1;    // 0..3
  const int wn   = wid & 1;     // 0..1
  const int lr   = lane & 15;   // row (A) / col (B) within 16
  const int kh   = lane >> 4;   // K half select (ISA 16-bit operand layout)

  const v8f vzero = {0.f, 0.f, 0.f, 0.f, 0.f, 0.f, 0.f, 0.f};
  v8f acc[2][4];
#pragma unroll
  for (int mt = 0; mt < 2; ++mt)
#pragma unroll
    for (int nt = 0; nt < 4; ++nt) acc[mt][nt] = vzero;

#pragma unroll
  for (int ks = 0; ks < KDIM; ks += 32) {
    const int k0 = ks + kh * 8;        // elems 0..7  -> K = ks + kh*8 + 0..7
    const int k1 = ks + 16 + kh * 8;   // elems 8..15 -> K = ks + 16 + kh*8 + 0..7

    v16bf ah[2], al[2];
#pragma unroll
    for (int mt = 0; mt < 2; ++mt) {
      const int row = (wm * 32 + mt * 16 + lr) * LDK;
      const v8bf h0 = *(const v8bf*)&sAh[row + k0];
      const v8bf h1 = *(const v8bf*)&sAh[row + k1];
      const v8bf l0 = *(const v8bf*)&sAl[row + k0];
      const v8bf l1 = *(const v8bf*)&sAl[row + k1];
      ah[mt] = __builtin_shufflevector(h0, h1, 0,1,2,3,4,5,6,7,8,9,10,11,12,13,14,15);
      al[mt] = __builtin_shufflevector(l0, l1, 0,1,2,3,4,5,6,7,8,9,10,11,12,13,14,15);
    }

#pragma unroll
    for (int nt = 0; nt < 4; ++nt) {
      const int col = (wn * 64 + nt * 16 + lr) * LDK;  // B built like A (Gram)
      const v8bf h0 = *(const v8bf*)&sBh[col + k0];
      const v8bf h1 = *(const v8bf*)&sBh[col + k1];
      const v8bf l0 = *(const v8bf*)&sBl[col + k0];
      const v8bf l1 = *(const v8bf*)&sBl[col + k1];
      const v16bf bh = __builtin_shufflevector(h0, h1, 0,1,2,3,4,5,6,7,8,9,10,11,12,13,14,15);
      const v16bf bl = __builtin_shufflevector(l0, l1, 0,1,2,3,4,5,6,7,8,9,10,11,12,13,14,15);
#pragma unroll
      for (int mt = 0; mt < 2; ++mt) {
        // 3-term bf16 split: ah*bh + al*bh + ah*bl  (drop al*bl, ~2^-16 rel)
        acc[mt][nt] = __builtin_amdgcn_wmma_f32_16x16x32_bf16(
            false, ah[mt], false, bh, (short)0, acc[mt][nt], false, false);
        acc[mt][nt] = __builtin_amdgcn_wmma_f32_16x16x32_bf16(
            false, al[mt], false, bh, (short)0, acc[mt][nt], false, false);
        acc[mt][nt] = __builtin_amdgcn_wmma_f32_16x16x32_bf16(
            false, ah[mt], false, bl, (short)0, acc[mt][nt], false, false);
      }
    }
  }

  // Epilogue: D-fragment layout (ISA): lane n = lane&15, VGPR r holds m = r + 8*(lane>>4).
  float lsum = 0.0f;
#pragma unroll
  for (int mt = 0; mt < 2; ++mt) {
    const int il0 = wm * 32 + mt * 16 + kh * 8;
#pragma unroll
    for (int nt = 0; nt < 4; ++nt) {
      const int jl  = wn * 64 + nt * 16 + lr;
      const int j   = jbase + jl;
      const float sqj = sSqB[jl];
#pragma unroll
      for (int r = 0; r < 8; ++r) {
        const int il = il0 + r;
        const int i  = ibase + il;
        const float g  = acc[mt][nt][r];
        const float d2 = sSqA[il] + sqj - 2.0f * g;
        const float dd = (d2 > 0.0f) ? __builtin_amdgcn_sqrtf(d2) : 0.0f;
        const float Dij = __builtin_nontemporal_load(&D[(size_t)i * NPTS + j]);
        const float den = Dij + ((i == j) ? 1.0f : 0.0f);
        lsum += fabsf(dd - Dij) * __builtin_amdgcn_rcpf(den);
      }
    }
  }
  if (bx != by) lsum *= 2.0f;   // symmetric contribution of the mirrored tile

  // wave32 shuffle reduction, then block reduction, one atomic per block
#pragma unroll
  for (int off = 16; off > 0; off >>= 1)
    lsum += __shfl_xor(lsum, off, 32);
  if (lane == 0) sRed[wid] = lsum;
  __syncthreads();
  if (tid == 0) {
    float tot = 0.0f;
#pragma unroll
    for (int w = 0; w < 8; ++w) tot += sRed[w];
    const float inv_denom = 1.0f / ((float)NPTS * (float)NPTS - (float)NPTS);
    atomicAdd(out, tot * inv_denom);
  }
}

extern "C" void kernel_launch(void* const* d_in, const int* in_sizes, int n_in,
                              void* d_out, int out_size, void* d_ws, size_t ws_size,
                              hipStream_t stream) {
  (void)in_sizes; (void)n_in; (void)out_size; (void)ws_size;
  const float* mapping = (const float*)d_in[0];
  const float* D       = (const float*)d_in[1];
  float* out = (float*)d_out;

  // Workspace layout: [sq: 8192 f32 = 32KB][hi: N*K bf16 = 2MB][lo: 2MB] (~4.03MB total)
  float* sq = (float*)d_ws;
  bf16*  hi = (bf16*)((char*)d_ws + 32768);
  bf16*  lo = (bf16*)((char*)d_ws + 32768 + (size_t)NPTS * KDIM * sizeof(bf16));

  dist_zero_kernel<<<1, 1, 0, stream>>>(out);
  dist_prep_kernel<<<NPTS, KDIM, 0, stream>>>(mapping, sq, hi, lo);

  const int nblocks = NT * (NT + 1) / 2;   // 2080 upper-triangular tiles
  dist_main_kernel<<<nblocks, 256, 0, stream>>>(D, sq, hi, lo, out);
}
